// GraphAttentionLayer_11433202941966
// MI455X (gfx1250) — matrix operand order repaired and checked by reading, not compiled, not run on hardware
//
#include <hip/hip_runtime.h>
#include <hip/hip_bf16.h>

#define N_NODES 8192
#define IN_FEAT 256
#define OUT_FEAT 64
#define ALPHA 0.2f
#define NEG_BIG -1000000000000.0f

typedef __attribute__((ext_vector_type(16))) __bf16 v16bf;
typedef __attribute__((ext_vector_type(8)))  float  v8f;

// ---------------------------------------------------------------------------
// Kernel 1: h = inp @ W   (8192x256 @ 256x64). Store f32 (for a-dots) and
// bf16 (WMMA B operand). ~0.27 GFLOP, negligible vs main kernel.
// ---------------------------------------------------------------------------
__global__ __launch_bounds__(256)
void gat_h_kernel(const float* __restrict__ inp,
                  const float* __restrict__ W,
                  float* __restrict__ hf,
                  __bf16* __restrict__ hb) {
    int gid = blockIdx.x * blockDim.x + threadIdx.x;   // 0 .. 8192*64-1
    int n = gid >> 6;
    int o = gid & 63;
    const float* irow = inp + (size_t)n * IN_FEAT;
    float acc = 0.f;
#pragma unroll 8
    for (int k = 0; k < IN_FEAT; ++k)
        acc = fmaf(irow[k], W[k * OUT_FEAT + o], acc);
    hf[gid] = acc;
    hb[gid] = (__bf16)acc;
}

// ---------------------------------------------------------------------------
// Kernel 2: f_src[n] = h[n]·a[0:64], f_dst[n] = h[n]·a[64:128]
// ---------------------------------------------------------------------------
__global__ __launch_bounds__(256)
void gat_f_kernel(const float* __restrict__ hf,
                  const float* __restrict__ a,
                  float* __restrict__ fsrc,
                  float* __restrict__ fdst) {
    int n = blockIdx.x * blockDim.x + threadIdx.x;
    const float* hrow = hf + (size_t)n * OUT_FEAT;
    float s = 0.f, d = 0.f;
#pragma unroll
    for (int o = 0; o < OUT_FEAT; ++o) {
        float h = hrow[o];
        s = fmaf(h, a[o], s);
        d = fmaf(h, a[OUT_FEAT + o], d);
    }
    fsrc[n] = s;
    fdst[n] = d;
}

// ---------------------------------------------------------------------------
// Kernel 3: fused masked-softmax attention + P@h via WMMA (flash style).
// One block = 16 rows; 8 waves split the 8192 columns into interleaved
// 32-wide tiles; partials merged through LDS at the end.
// ---------------------------------------------------------------------------
__global__ __launch_bounds__(256)
void gat_attn_kernel(const int*   __restrict__ adj,
                     const float* __restrict__ fsrc,
                     const float* __restrict__ fdst,
                     const __bf16* __restrict__ hb,
                     float* __restrict__ out) {
    __shared__ float lds_m[8][16];
    __shared__ float lds_l[8][16];
    __shared__ float lds_acc[8][16][OUT_FEAT];   // 32 KB
    __shared__ float fin_m[16];
    __shared__ float fin_l[16];

    const int w     = threadIdx.x >> 5;   // wave 0..7
    const int lane  = threadIdx.x & 31;
    const int rowL  = lane & 15;          // A-matrix row this lane owns
    const int rb    = blockIdx.x * 16;    // first global row of this block
    const int rowG  = rb + rowL;
    const int kbase = (lane < 16) ? 0 : 8; // A layout: K split per half-wave

    const float fs = fsrc[rowG];
    const int* adjRow = adj + (size_t)rowG * N_NODES;

    float m    = -__builtin_inff();
    float lsum = 0.f;
    v8f c0 = {}, c1 = {}, c2 = {}, c3 = {};   // 16x64 f32 accum (4 WMMA tiles)

    // 256 column tiles of 32; wave w takes every 8th tile.
#pragma unroll 1
    for (int jt = w; jt < N_NODES / 32; jt += 8) {
        const int j0 = jt * 32;

        // This lane's 16 K positions: K = kbase+t (t<8), K = kbase+16+(t-8)
        const int*   ap0 = adjRow + j0 + kbase;
        const int*   ap1 = adjRow + j0 + kbase + 16;
        const float* dp0 = fdst   + j0 + kbase;
        const float* dp1 = fdst   + j0 + kbase + 16;

        int4   a0 = *(const int4*)(ap0);
        int4   a1 = *(const int4*)(ap0 + 4);
        int4   a2 = *(const int4*)(ap1);
        int4   a3 = *(const int4*)(ap1 + 4);
        float4 d0 = *(const float4*)(dp0);
        float4 d1 = *(const float4*)(dp0 + 4);
        float4 d2 = *(const float4*)(dp1);
        float4 d3 = *(const float4*)(dp1 + 4);

        int   av[16] = {a0.x,a0.y,a0.z,a0.w, a1.x,a1.y,a1.z,a1.w,
                        a2.x,a2.y,a2.z,a2.w, a3.x,a3.y,a3.z,a3.w};
        float fd[16] = {d0.x,d0.y,d0.z,d0.w, d1.x,d1.y,d1.z,d1.w,
                        d2.x,d2.y,d2.z,d2.w, d3.x,d3.y,d3.z,d3.w};

        float e[16];
        float tmax = -__builtin_inff();
#pragma unroll
        for (int t = 0; t < 16; ++t) {
            float x = fs + fd[t];
            x = (x >= 0.f) ? x : ALPHA * x;       // leaky relu
            x = (av[t] > 0) ? x : NEG_BIG;        // adjacency mask
            e[t] = x;
            tmax = fmaxf(tmax, x);
        }

        // Row max over the full 32-col tile (partner lane holds other half)
        float pmax = __shfl_xor(tmax, 16, 32);
        float mnew = fmaxf(m, fmaxf(tmax, pmax));

        float psum = 0.f;
        v16bf A;
#pragma unroll
        for (int t = 0; t < 16; ++t) {
            float p = __expf(e[t] - mnew);
            psum += p;
            A[t] = (__bf16)p;                     // pack P tile (A operand)
        }
        psum += __shfl_xor(psum, 16, 32);

        float scale = __expf(m - mnew);           // first iter: exp(-inf)=0
        lsum = lsum * scale + psum;
        m = mnew;

        // Rescale accumulators: C layout puts rows 0-7 (lanes<16) / 8-15
        // (lanes>=16) in VGPRs 0-7; fetch per-row scale from owning lane.
#pragma unroll
        for (int i = 0; i < 8; ++i) {
            float si = __shfl(scale, (lane < 16) ? i : (8 + i), 32);
            c0[i] *= si; c1[i] *= si; c2[i] *= si; c3[i] *= si;
        }

        // B operand: lane = N col (feat), elements = K rows in A's K order.
        v16bf B0, B1, B2, B3;
#pragma unroll
        for (int t = 0; t < 16; ++t) {
            const int kk = kbase + ((t < 8) ? t : (t + 8));
            const __bf16* hrow = hb + (size_t)(j0 + kk) * OUT_FEAT + rowL;
            B0[t] = hrow[0];
            B1[t] = hrow[16];
            B2[t] = hrow[32];
            B3[t] = hrow[48];
        }

        c0 = __builtin_amdgcn_wmma_f32_16x16x32_bf16(false, A, false, B0,
                                                     (short)0, c0, false, false);
        c1 = __builtin_amdgcn_wmma_f32_16x16x32_bf16(false, A, false, B1,
                                                     (short)0, c1, false, false);
        c2 = __builtin_amdgcn_wmma_f32_16x16x32_bf16(false, A, false, B2,
                                                     (short)0, c2, false, false);
        c3 = __builtin_amdgcn_wmma_f32_16x16x32_bf16(false, A, false, B3,
                                                     (short)0, c3, false, false);
    }

    // ---- merge the 8 per-wave partials through LDS ----
    if (lane < 16) { lds_m[w][lane] = m; lds_l[w][lane] = lsum; }
    const int colB  = lane & 15;
    const int rbase = (lane < 16) ? 0 : 8;
#pragma unroll
    for (int i = 0; i < 8; ++i) {
        lds_acc[w][rbase + i][colB +  0] = c0[i];
        lds_acc[w][rbase + i][colB + 16] = c1[i];
        lds_acc[w][rbase + i][colB + 32] = c2[i];
        lds_acc[w][rbase + i][colB + 48] = c3[i];
    }
    __syncthreads();

    if (threadIdx.x < 16) {
        int r = threadIdx.x;
        float M = -__builtin_inff();
#pragma unroll
        for (int ww = 0; ww < 8; ++ww) M = fmaxf(M, lds_m[ww][r]);
        float L = 0.f;
#pragma unroll
        for (int ww = 0; ww < 8; ++ww)
            L += lds_l[ww][r] * __expf(lds_m[ww][r] - M);
        fin_m[r] = M;
        fin_l[r] = L;
    }
    __syncthreads();

    // 16 rows x 64 feats = 1024 outputs; 4 per thread. Normalize + ELU.
#pragma unroll
    for (int q = 0; q < 4; ++q) {
        int idx  = threadIdx.x + 256 * q;
        int r    = idx >> 6;
        int cidx = idx & 63;
        float M = fin_m[r];
        float v = 0.f;
#pragma unroll
        for (int ww = 0; ww < 8; ++ww)
            v += lds_acc[ww][r][cidx] * __expf(lds_m[ww][r] - M);
        v /= fin_l[r];
        v = (v > 0.f) ? v : (expf(v) - 1.f);      // ELU (alpha=1)
        out[(size_t)(rb + r) * OUT_FEAT + cidx] = v;
    }
}

// ---------------------------------------------------------------------------
extern "C" void kernel_launch(void* const* d_in, const int* in_sizes, int n_in,
                              void* d_out, int out_size, void* d_ws, size_t ws_size,
                              hipStream_t stream) {
    const float* inp = (const float*)d_in[0];   // (1, 8192, 256) f32
    const int*   adj = (const int*)d_in[1];     // (1, 8192, 8192) i32
    const float* W   = (const float*)d_in[2];   // (1, 256, 64) f32
    const float* a   = (const float*)d_in[3];   // (128, 1) f32
    float* out = (float*)d_out;                 // (1, 8192, 64) f32

    // Workspace layout (~3.2 MB): h_f32 | h_bf16 | f_src | f_dst
    char* ws = (char*)d_ws;
    float*  hf   = (float*)(ws);
    __bf16* hb   = (__bf16*)(ws + (size_t)N_NODES * OUT_FEAT * 4);
    float*  fsrc = (float*)(ws + (size_t)N_NODES * OUT_FEAT * 6);
    float*  fdst = fsrc + N_NODES;

    gat_h_kernel<<<(N_NODES * OUT_FEAT) / 256, 256, 0, stream>>>(inp, W, hf, hb);
    gat_f_kernel<<<N_NODES / 256, 256, 0, stream>>>(hf, a, fsrc, fdst);
    gat_attn_kernel<<<N_NODES / 16, 256, 0, stream>>>(adj, fsrc, fdst, hb, out);
}
